// JointEuclideanProbe_90993177133168
// MI455X (gfx1250) — compile-verified, matrix operand order checked by reading, not run on hardware
//
#include <hip/hip_runtime.h>
#include <hip/hip_bf16.h>

// Problem dims
#define BB 8
#define SS 512
#define DD 768
#define RR 128
#define MROWS (BB * SS)   // 4096

typedef __attribute__((ext_vector_type(16))) __bf16 v16bf;
typedef __attribute__((ext_vector_type(8)))  __bf16 v8bf;
typedef __attribute__((ext_vector_type(8)))  float  v8f;
typedef __attribute__((ext_vector_type(4)))  float  v4f;

// A/B fragment packing for v_wmma_f32_16x16x32_bf16 (wave32):
//   lanes 0-15 : elements 0..7 = K 0..7,  elements 8..15 = K 16..23
//   lanes 16-31: elements 0..7 = K 8..15, elements 8..15 = K 24..31
// Caller passes a pointer already offset by the per-lane K base (koff),
// so we always read [0..7] and [16..23] relative to it.

__device__ inline v16bf frag_from_f32(const float* __restrict__ p) {
    v4f a0 = *(const v4f*)(p);
    v4f a1 = *(const v4f*)(p + 4);
    v4f b0 = *(const v4f*)(p + 16);
    v4f b1 = *(const v4f*)(p + 20);
    v16bf r;
#pragma unroll
    for (int t = 0; t < 4; ++t) {
        r[t]      = (__bf16)a0[t];
        r[4 + t]  = (__bf16)a1[t];
        r[8 + t]  = (__bf16)b0[t];
        r[12 + t] = (__bf16)b1[t];
    }
    return r;
}

__device__ inline v16bf frag_from_bf16(const __bf16* __restrict__ p) {
    v8bf lo = *(const v8bf*)(p);
    v8bf hi = *(const v8bf*)(p + 16);
    return __builtin_shufflevector(lo, hi, 0, 1, 2, 3, 4, 5, 6, 7,
                                           8, 9, 10, 11, 12, 13, 14, 15);
}

// Kernel 1: projection GEMM. blockIdx.y == 0 -> proj_dist (store P bf16 + norms),
// blockIdx.y == 1 -> proj_depth (write depths = row squared norms directly).
// Each wave: 16 rows of E x all 128 columns of W^T, K = 768 in steps of 32.
__global__ __launch_bounds__(256) void proj_kernel(
    const float* __restrict__ E,    // (4096, 768)
    const float* __restrict__ Wd,   // (128, 768)
    const float* __restrict__ Wz,   // (128, 768)
    __bf16* __restrict__ Pd,        // (4096, 128) workspace, bf16
    float* __restrict__ norms,      // (4096) workspace
    float* __restrict__ depths)     // (4096) output slice
{
    const int lane  = threadIdx.x & 31;
    const int wave  = threadIdx.x >> 5;
    const int mbase = (blockIdx.x * 8 + wave) * 16;      // 16-row strip
    const bool depth_mode = (blockIdx.y == 1);
    const float* __restrict__ W = depth_mode ? Wz : Wd;

    const int row  = mbase + (lane & 15);
    const int koff = (lane >> 4) * 8;
    const int ncol = lane & 15;

    v8f acc[8] = {};

    for (int kt = 0; kt < DD / 32; ++kt) {
        const int k0 = kt * 32 + koff;
        v16bf a = frag_from_f32(&E[(size_t)row * DD + k0]);
#pragma unroll
        for (int nt = 0; nt < 8; ++nt) {
            v16bf b = frag_from_f32(&W[(size_t)(nt * 16 + ncol) * DD + k0]);
            acc[nt] = __builtin_amdgcn_wmma_f32_16x16x32_bf16(
                false, a, false, b, (short)0, acc[nt], false, false);
        }
    }

    // Per-lane partial row sums of squares: VGPR v holds row mbase+v (lanes 0-15)
    // or mbase+8+v (lanes 16-31), column nt*16 + (lane&15).
    float sv[8];
#pragma unroll
    for (int v = 0; v < 8; ++v) {
        float s = 0.0f;
#pragma unroll
        for (int nt = 0; nt < 8; ++nt) {
            float c = acc[nt][v];
            s += c * c;
        }
        sv[v] = s;
    }
    // Reduce across the 16 lanes of each half (xor 1,2,4,8 stays within half).
#pragma unroll
    for (int v = 0; v < 8; ++v) {
        float s = sv[v];
        s += __shfl_xor(s, 1, 32);
        s += __shfl_xor(s, 2, 32);
        s += __shfl_xor(s, 4, 32);
        s += __shfl_xor(s, 8, 32);
        sv[v] = s;
    }

    if (depth_mode) {
        if (lane == 0 || lane == 16) {
            const int mo = mbase + (lane >> 4) * 8;
#pragma unroll
            for (int v = 0; v < 8; ++v) depths[mo + v] = sv[v];
        }
    } else {
        if (lane == 0 || lane == 16) {
            const int mo = mbase + (lane >> 4) * 8;
#pragma unroll
            for (int v = 0; v < 8; ++v) norms[mo + v] = sv[v];
        }
        // Store P as bf16 for the Gram-matrix kernel.
        const int rowoff = (lane >> 4) * 8;
#pragma unroll
        for (int nt = 0; nt < 8; ++nt) {
#pragma unroll
            for (int v = 0; v < 8; ++v) {
                const int m = mbase + rowoff + v;
                Pd[(size_t)m * RR + nt * 16 + ncol] = (__bf16)acc[nt][v];
            }
        }
    }
}

// Kernel 2: per-batch Gram matrix -> distances. One 16x16 tile per wave,
// K = 128 -> 4 x wmma 16x16x32. dist[i,j] = n_i + n_j - 2 * <p_i, p_j>.
__global__ __launch_bounds__(256) void dist_kernel(
    const __bf16* __restrict__ P,   // (4096, 128) bf16
    const float* __restrict__ norms,// (4096)
    float* __restrict__ dist)       // (8, 512, 512)
{
    const int lane = threadIdx.x & 31;
    const int wave = threadIdx.x >> 5;
    const int tile = blockIdx.x * 8 + wave;   // 8192 tiles total
    const int b  = tile >> 10;                // 1024 tiles (32x32) per batch
    const int t  = tile & 1023;
    const int it = t >> 5;
    const int jt = t & 31;

    const int koff = (lane >> 4) * 8;
    const int ncol = lane & 15;
    const int arow = b * SS + it * 16 + ncol; // A: rows i
    const int brow = b * SS + jt * 16 + ncol; // B: column n = row j of P

    v8f acc = {};
#pragma unroll
    for (int kt = 0; kt < RR / 32; ++kt) {
        const int k0 = kt * 32 + koff;
        v16bf a = frag_from_bf16(&P[(size_t)arow * RR + k0]);
        v16bf bb = frag_from_bf16(&P[(size_t)brow * RR + k0]);
        acc = __builtin_amdgcn_wmma_f32_16x16x32_bf16(
            false, a, false, bb, (short)0, acc, false, false);
    }

    const float nj = norms[b * SS + jt * 16 + ncol];
    const int rowoff = (lane >> 4) * 8;
#pragma unroll
    for (int v = 0; v < 8; ++v) {
        const int i = it * 16 + rowoff + v;
        const float ni = norms[b * SS + i];
        const float dv = ni + nj - 2.0f * acc[v];
        dist[((size_t)b * SS + i) * SS + jt * 16 + ncol] = dv;
    }
}

extern "C" void kernel_launch(void* const* d_in, const int* in_sizes, int n_in,
                              void* d_out, int out_size, void* d_ws, size_t ws_size,
                              hipStream_t stream) {
    const float* E  = (const float*)d_in[0];   // embeddings (8,512,768)
    const float* Wd = (const float*)d_in[1];   // proj_dist  (128,768)
    const float* Wz = (const float*)d_in[2];   // proj_depth (128,768)

    float* out    = (float*)d_out;
    float* dist   = out;                                   // (8,512,512)
    float* depths = out + (size_t)BB * SS * SS;            // (8,512)

    __bf16* Pd   = (__bf16*)d_ws;                          // 4096*128*2 = 1 MB
    float*  norms = (float*)((char*)d_ws + (size_t)MROWS * RR * sizeof(__bf16));

    // 256 m-strips of 16 rows; 8 waves/block -> 32 blocks; y: {dist, depth}
    proj_kernel<<<dim3(32, 2), 256, 0, stream>>>(E, Wd, Wz, Pd, norms, depths);

    // 8192 16x16 tiles / 8 waves per block = 1024 blocks
    dist_kernel<<<1024, 256, 0, stream>>>(Pd, norms, dist);
}
